// GraphResidualBlock_25623774888137
// MI455X (gfx1250) — matrix operand order repaired and checked by reading, not compiled, not run on hardware
//
#include <hip/hip_runtime.h>

// ---------------------------------------------------------------------------
// Types for CDNA5 WMMA (wave32): bf16 A/B fragments, f32 accumulator.
// ---------------------------------------------------------------------------
typedef __attribute__((ext_vector_type(16))) __bf16 v16bf;
typedef __attribute__((ext_vector_type(8)))  __bf16 v8bf;
typedef __attribute__((ext_vector_type(8)))  float  v8f;

#define LDS_STRIDE 136   // 128 + 8 pad (keeps 16B alignment, breaks bank conflicts)

// Flags for the generic GEMM
#define F_LN    1   // A <- (A - mu[col]) * sdinv[col]  (LayerNorm over node dim)
#define F_RELU  2   // A <- max(A, 0)
#define F_OBF16 4   // OUT stored as bf16 instead of f32

// ---------------------------------------------------------------------------
// OUT[M,128] = op(A[M,128]) @ W^T[128,128] + bias + addin
// W is torch-layout [out,in]: OUT[m,n] = sum_k A[m,k] * W[n,k]
// Block: 256 threads = 8 waves. Block tile: 64 rows x 128 cols.
// Wave w owns columns 16w..16w+15; iterates 4 row tiles x 4 K-steps
// => 16 x v_wmma_f32_16x16x32_bf16 per wave.
// ---------------------------------------------------------------------------
__global__ __launch_bounds__(256) void gemm128_wmma(
    const float* __restrict__ A, const float* __restrict__ W,
    const float* __restrict__ bias, const float* __restrict__ addin,
    const float* __restrict__ mu, const float* __restrict__ sdinv,
    void* __restrict__ OUTv, long M, int flags)
{
    __shared__ __bf16 sW[128 * LDS_STRIDE];  // [n][k], bf16
    __shared__ __bf16 sA[64  * LDS_STRIDE];  // [m][k], bf16 (pre-ops applied)

    const int  tid  = threadIdx.x;
    const int  wave = tid >> 5;
    const int  lane = tid & 31;
    const long row0 = (long)blockIdx.x * 64;

    // Stage weights (fp32 -> bf16), coalesced.
    for (int i = tid; i < 128 * 128; i += 256) {
        int r = i >> 7, c = i & 127;
        sW[r * LDS_STRIDE + c] = (__bf16)W[i];
    }
    // Stage A tile with fused pre-ops.
    for (int i = tid; i < 64 * 128; i += 256) {
        int  r = i >> 7, c = i & 127;
        long row = row0 + r;
        float a = (row < M) ? A[row * 128 + c] : 0.0f;
        if (flags & F_LN)   a = (a - mu[c]) * sdinv[c];
        if (flags & F_RELU) a = fmaxf(a, 0.0f);
        sA[r * LDS_STRIDE + c] = (__bf16)a;
    }
    __syncthreads();

    v8f acc[4] = {{}, {}, {}, {}};
    const int mrow = lane & 15;
    const int ncol = (wave << 4) + (lane & 15);
    const int g    = (lane < 16) ? 0 : 1;   // lane-half selects K sub-blocks

#pragma unroll
    for (int k0 = 0; k0 < 128; k0 += 32) {
        // B fragment: lanes 0-15 hold K = k0+0..15, lanes 16-31 K = k0+16..31
        const v8bf* pb = (const v8bf*)&sW[ncol * LDS_STRIDE + k0];
        v8bf b0 = pb[2 * g];
        v8bf b1 = pb[2 * g + 1];
        v16bf bfrag = __builtin_shufflevector(b0, b1,
            0, 1, 2, 3, 4, 5, 6, 7, 8, 9, 10, 11, 12, 13, 14, 15);
#pragma unroll
        for (int mt = 0; mt < 4; ++mt) {
            // A fragment: lanes 0-15 hold K = k0+0..15,
            //             lanes 16-31 K = k0+16..31
            const v8bf* pa = (const v8bf*)&sA[(mt * 16 + mrow) * LDS_STRIDE + k0];
            v8bf a0 = pa[2 * g];
            v8bf a1 = pa[2 * g + 1];
            v16bf afrag = __builtin_shufflevector(a0, a1,
                0, 1, 2, 3, 4, 5, 6, 7, 8, 9, 10, 11, 12, 13, 14, 15);
            acc[mt] = __builtin_amdgcn_wmma_f32_16x16x32_bf16(
                false, afrag, false, bfrag, (short)0, acc[mt], false, false);
        }
    }

    // C/D layout: VGPR r holds M = r + (lane<16 ? 0 : 8), N = lane%16.
    const float bb    = bias ? bias[ncol] : 0.0f;
    const int   mbase = (lane < 16) ? 0 : 8;
#pragma unroll
    for (int mt = 0; mt < 4; ++mt) {
#pragma unroll
        for (int r = 0; r < 8; ++r) {
            long row = row0 + mt * 16 + mbase + r;
            if (row < M) {
                float val = acc[mt][r] + bb;
                if (addin) val += addin[row * 128 + ncol];
                if (flags & F_OBF16)
                    ((__bf16*)OUTv)[row * 128 + ncol] = (__bf16)val;
                else
                    ((float*)OUTv)[row * 128 + ncol] = val;
            }
        }
    }
}

// ---------------------------------------------------------------------------
// Float atomic max via sign-aware integer min/max (valid with -inf init).
// ---------------------------------------------------------------------------
__device__ __forceinline__ void atomicMaxFloat(float* addr, float val)
{
    if (val >= 0.0f)
        atomicMax((int*)addr, __float_as_int(val));
    else
        atomicMin((unsigned int*)addr, (unsigned int)__float_as_int(val));
}

__global__ void init_ms_kernel(float* __restrict__ m, float* __restrict__ s, long n)
{
    long i = (long)blockIdx.x * blockDim.x + threadIdx.x;
    if (i < n) { m[i] = -__builtin_huge_valf(); s[i] = 0.0f; }
}

// ---------------------------------------------------------------------------
// One wave per edge: lane = channel c (0..31). For each head h:
//   alpha[e,h] = dot(q[dst,h,:], k[src,h,:] + e[e,h,:]) / sqrt(32)
// then scatter-max into mmax[dst,h].
// ---------------------------------------------------------------------------
__global__ __launch_bounds__(256) void edge_alpha_kernel(
    const float* __restrict__ q, const float* __restrict__ k,
    const __bf16* __restrict__ eproj, const int* __restrict__ src,
    const int* __restrict__ dst, float* __restrict__ alpha,
    float* __restrict__ mmax, long E)
{
    long edge = (long)blockIdx.x * 8 + (threadIdx.x >> 5);
    if (edge >= E) return;
    int  lane = threadIdx.x & 31;
    long s = src[edge], d = dst[edge];
    const float scale = 0.17677669529663687f;  // 1/sqrt(32)
#pragma unroll
    for (int h = 0; h < 4; ++h) {
        int   off = h * 32 + lane;
        float val = q[d * 128 + off] *
                    (k[s * 128 + off] + (float)eproj[edge * 128 + off]);
#pragma unroll
        for (int w = 16; w > 0; w >>= 1) val += __shfl_xor(val, w, 32);
        if (lane == 0) {
            float a = val * scale;
            alpha[edge * 4 + h] = a;
            atomicMaxFloat(&mmax[d * 4 + h], a);
        }
    }
}

// alpha <- exp(alpha - m[dst]); s[dst] += alpha     (one thread per edge*head)
__global__ void edge_expsum_kernel(
    float* __restrict__ alpha, const float* __restrict__ mmax,
    const int* __restrict__ dst, float* __restrict__ ssum, long E4)
{
    long i = (long)blockIdx.x * blockDim.x + threadIdx.x;
    if (i >= E4) return;
    long edge = i >> 2;
    int  h    = (int)(i & 3);
    long d    = dst[edge];
    float a = __expf(alpha[i] - mmax[d * 4 + h]);
    alpha[i] = a;
    atomicAdd(&ssum[d * 4 + h], a);
}

// agg[dst] += (v[src] + e) * alpha / s[dst]   (wave per edge; agg preloaded
// with the skip projection so out = agg + skip comes for free)
__global__ __launch_bounds__(256) void edge_agg_kernel(
    const float* __restrict__ v, const __bf16* __restrict__ eproj,
    const float* __restrict__ alpha, const float* __restrict__ ssum,
    const int* __restrict__ src, const int* __restrict__ dst,
    float* __restrict__ agg, long E)
{
    long edge = (long)blockIdx.x * 8 + (threadIdx.x >> 5);
    if (edge >= E) return;
    int  lane = threadIdx.x & 31;
    long s = src[edge], d = dst[edge];
#pragma unroll
    for (int h = 0; h < 4; ++h) {
        float wgt = alpha[edge * 4 + h] / ssum[d * 4 + h];
        int   off = h * 32 + lane;
        float msg = (v[s * 128 + off] + (float)eproj[edge * 128 + off]) * wgt;
        atomicAdd(&agg[d * 128 + off], msg);
    }
}

// ---------------------------------------------------------------------------
// Per-feature mean / 1/(std+eps) over the node dimension (ddof = 0).
// One block per column.
// ---------------------------------------------------------------------------
__global__ __launch_bounds__(256) void colstats_kernel(
    const float* __restrict__ h, float* __restrict__ mu,
    float* __restrict__ sdinv, long N)
{
    __shared__ float ssum[256];
    __shared__ float ssq[256];
    int j = blockIdx.x;
    int t = threadIdx.x;
    float s = 0.0f, q = 0.0f;
    for (long n = t; n < N; n += 256) {
        float v = h[n * 128 + j];
        s += v;
        q += v * v;
    }
    ssum[t] = s; ssq[t] = q;
    __syncthreads();
    for (int w = 128; w > 0; w >>= 1) {
        if (t < w) { ssum[t] += ssum[t + w]; ssq[t] += ssq[t + w]; }
        __syncthreads();
    }
    if (t == 0) {
        float m   = ssum[0] / (float)N;
        float var = ssq[0] / (float)N - m * m;
        var = var > 0.0f ? var : 0.0f;
        mu[j]    = m;
        sdinv[j] = 1.0f / (sqrtf(var) + 1e-5f);
    }
}

// ---------------------------------------------------------------------------
// Launcher
// ---------------------------------------------------------------------------
extern "C" void kernel_launch(void* const* d_in, const int* in_sizes, int n_in,
                              void* d_out, int out_size, void* d_ws, size_t ws_size,
                              hipStream_t stream)
{
    const float* x     = (const float*)d_in[0];
    const int*   ei    = (const int*)  d_in[1];
    const float* ea    = (const float*)d_in[2];
    const float* Wq    = (const float*)d_in[3];
    const float* bq    = (const float*)d_in[4];
    const float* Wk    = (const float*)d_in[5];
    const float* bk    = (const float*)d_in[6];
    const float* Wv    = (const float*)d_in[7];
    const float* bv    = (const float*)d_in[8];
    const float* We    = (const float*)d_in[9];
    const float* Wskip = (const float*)d_in[10];
    const float* bskip = (const float*)d_in[11];
    const float* Wp    = (const float*)d_in[12];
    const float* bp    = (const float*)d_in[13];
    const float* Wm    = (const float*)d_in[14];
    const float* bm    = (const float*)d_in[15];

    const long N = (long)in_sizes[0] / 128;
    const long E = (long)in_sizes[1] / 2;
    const int* src = ei;
    const int* dst = ei + E;

    // Carve workspace (fp32 buffers first, bf16 eproj last).
    float* w = (float*)d_ws;
    float* qb    = w; w += N * 128;
    float* kb    = w; w += N * 128;
    float* vb    = w; w += N * 128;
    float* agg   = w; w += N * 128;   // skip projection, then += messages
    float* hbuf  = w; w += N * 128;   // post-physics activations
    float* mbuf  = w; w += N * 4;     // per (node, head) running max
    float* sbuf  = w; w += N * 4;     // per (node, head) exp-sum
    float* mu    = w; w += 128;
    float* sdinv = w; w += 128;
    float* alpha = w; w += E * 4;     // scores -> exp(scores - max)
    __bf16* eproj = (__bf16*)w;       // [E,128] bf16 edge projections

    const dim3 blk(256);
    const unsigned gN = (unsigned)((N + 63) / 64);
    const unsigned gE = (unsigned)((E + 63) / 64);

    // Node projections q / k / v / skip (skip lands directly in agg).
    gemm128_wmma<<<gN, blk, 0, stream>>>(x, Wq, bq, nullptr, nullptr, nullptr, qb,  N, 0);
    gemm128_wmma<<<gN, blk, 0, stream>>>(x, Wk, bk, nullptr, nullptr, nullptr, kb,  N, 0);
    gemm128_wmma<<<gN, blk, 0, stream>>>(x, Wv, bv, nullptr, nullptr, nullptr, vb,  N, 0);
    gemm128_wmma<<<gN, blk, 0, stream>>>(x, Wskip, bskip, nullptr, nullptr, nullptr, agg, N, 0);

    // Edge projection e = edge_attr @ We^T (no bias), stored bf16.
    gemm128_wmma<<<gE, blk, 0, stream>>>(ea, We, nullptr, nullptr, nullptr, nullptr,
                                         (void*)eproj, E, F_OBF16);

    // Scatter-softmax attention.
    init_ms_kernel<<<(unsigned)((N * 4 + 255) / 256), blk, 0, stream>>>(mbuf, sbuf, N * 4);
    edge_alpha_kernel<<<(unsigned)((E + 7) / 8), blk, 0, stream>>>(
        qb, kb, eproj, src, dst, alpha, mbuf, E);
    edge_expsum_kernel<<<(unsigned)((E * 4 + 255) / 256), blk, 0, stream>>>(
        alpha, mbuf, dst, sbuf, E * 4);
    edge_agg_kernel<<<(unsigned)((E + 7) / 8), blk, 0, stream>>>(
        vb, eproj, alpha, sbuf, src, dst, agg, E);

    // h = relu(agg) @ Wp^T + bp
    gemm128_wmma<<<gN, blk, 0, stream>>>(agg, Wp, bp, nullptr, nullptr, nullptr,
                                         hbuf, N, F_RELU);
    // LayerNorm stats over the node dimension.
    colstats_kernel<<<128, blk, 0, stream>>>(hbuf, mu, sdinv, N);
    // out = relu((h - mu) / (sd + eps)) @ Wm^T + bm + x
    gemm128_wmma<<<gN, blk, 0, stream>>>(hbuf, Wm, bm, x, mu, sdinv,
                                         d_out, N, F_LN | F_RELU);
}
